// PointNetPlusSetAbstactionMSG_63874753626490
// MI455X (gfx1250) — matrix operand order repaired and checked by reading, not compiled, not run on hardware
//
#include <hip/hip_runtime.h>

// ---------------------------------------------------------------------------
// PointNet++ Set Abstraction (MSG) for MI455X / gfx1250  (wave32, WMMA bf16)
//   B=8, N=4096, C=64, S(=NPOINT)=1024
//   radii {0.1,0.2,0.4}, nsample {16,32,128}
//   MLPs: [32,32,64] [64,64,128] [64,96,128], cin = 67 padded to 96
// ---------------------------------------------------------------------------

typedef __attribute__((ext_vector_type(16))) __bf16 bf16x16;
typedef __attribute__((ext_vector_type(8)))  __bf16 bf16x8;
typedef __attribute__((ext_vector_type(8)))  float  f32x8;

#define NBATCH 8
#define NPTS   4096
#define NFEAT  64
#define NCTR   1024
#define FEAT_STRIDE 320

// ---------------------------------------------------------------------------
// 1) Farthest point sampling: one block per batch, dists live in registers.
// ---------------------------------------------------------------------------
__global__ __launch_bounds__(256) void fps_kernel(const float* __restrict__ xyz,
                                                  float* __restrict__ newxyz_ws,
                                                  float* __restrict__ out_newxyz) {
  const int b = blockIdx.x;
  const int t = threadIdx.x;
  __shared__ float sv[256];
  __shared__ int   si[256];
  __shared__ float spx, spy, spz;

  float px[16], py[16], pz[16], d[16];
  const float x0 = xyz[(b * NPTS) * 3 + 0];
  const float y0 = xyz[(b * NPTS) * 3 + 1];
  const float z0 = xyz[(b * NPTS) * 3 + 2];
#pragma unroll
  for (int k = 0; k < 16; ++k) {
    const int j = t + k * 256;
    px[k] = xyz[(b * NPTS + j) * 3 + 0];
    py[k] = xyz[(b * NPTS + j) * 3 + 1];
    pz[k] = xyz[(b * NPTS + j) * 3 + 2];
    const float dx = px[k] - x0, dy = py[k] - y0, dz = pz[k] - z0;
    d[k] = dx * dx + dy * dy + dz * dz;
  }
  if (t == 0) {
    newxyz_ws[(b * NCTR) * 3 + 0] = x0;  out_newxyz[(b * NCTR) * 3 + 0] = x0;
    newxyz_ws[(b * NCTR) * 3 + 1] = y0;  out_newxyz[(b * NCTR) * 3 + 1] = y0;
    newxyz_ws[(b * NCTR) * 3 + 2] = z0;  out_newxyz[(b * NCTR) * 3 + 2] = z0;
  }
  for (int it = 1; it < NCTR; ++it) {
    float bv = -1.0f; int bj = 0;
#pragma unroll
    for (int k = 0; k < 16; ++k) {           // strict > keeps smallest j on ties
      const int j = t + k * 256;
      if (d[k] > bv) { bv = d[k]; bj = j; }
    }
    sv[t] = bv; si[t] = bj;
    __syncthreads();
    for (int sft = 128; sft > 0; sft >>= 1) {
      if (t < sft) {
        const float ov = sv[t + sft]; const int oi = si[t + sft];
        if (ov > sv[t] || (ov == sv[t] && oi < si[t])) { sv[t] = ov; si[t] = oi; }
      }
      __syncthreads();
    }
    if (t == 0) {
      const int nxt = si[0];
      const float qx = xyz[(b * NPTS + nxt) * 3 + 0];
      const float qy = xyz[(b * NPTS + nxt) * 3 + 1];
      const float qz = xyz[(b * NPTS + nxt) * 3 + 2];
      spx = qx; spy = qy; spz = qz;
      newxyz_ws[(b * NCTR + it) * 3 + 0] = qx;  out_newxyz[(b * NCTR + it) * 3 + 0] = qx;
      newxyz_ws[(b * NCTR + it) * 3 + 1] = qy;  out_newxyz[(b * NCTR + it) * 3 + 1] = qy;
      newxyz_ws[(b * NCTR + it) * 3 + 2] = qz;  out_newxyz[(b * NCTR + it) * 3 + 2] = qz;
    }
    __syncthreads();
    const float qx = spx, qy = spy, qz = spz;
#pragma unroll
    for (int k = 0; k < 16; ++k) {
      const float dx = px[k] - qx, dy = py[k] - qy, dz = pz[k] - qz;
      const float dd = dx * dx + dy * dy + dz * dz;
      d[k] = d[k] < dd ? d[k] : dd;
    }
  }
}

// ---------------------------------------------------------------------------
// 2) Ball query: one wave per (b,s); ordered collection via ballot prefix.
// ---------------------------------------------------------------------------
template <int NSAMP>
__global__ __launch_bounds__(256) void qball_kernel(const float* __restrict__ xyz,
                                                    const float* __restrict__ newxyz,
                                                    int* __restrict__ idxout, float r2) {
  __shared__ unsigned bal[8];
  const int lane = threadIdx.x & 31;
  const int wave = threadIdx.x >> 5;
  const int bs = blockIdx.x * 8 + wave;
  const int b = bs >> 10;
  const float cx = newxyz[bs * 3 + 0];
  const float cy = newxyz[bs * 3 + 1];
  const float cz = newxyz[bs * 3 + 2];
  int* out = idxout + (size_t)bs * NSAMP;
  int cnt = 0;
  for (int base = 0; base < NPTS; base += 32) {
    const int j = base + lane;
    const float dx = xyz[(b * NPTS + j) * 3 + 0] - cx;
    const float dy = xyz[(b * NPTS + j) * 3 + 1] - cy;
    const float dz = xyz[(b * NPTS + j) * 3 + 2] - cz;
    const bool inball = (dx * dx + dy * dy + dz * dz) < r2;
#if __has_builtin(__builtin_amdgcn_ballot_w32)
    const unsigned m = __builtin_amdgcn_ballot_w32(inball);
#else
    if (lane == 0) bal[wave] = 0u;
    atomicOr(&bal[wave], inball ? (1u << lane) : 0u);
    const unsigned m = bal[wave];
#endif
    const int pos = cnt + __popc(m & ((1u << lane) - 1u));
    if (inball && pos < NSAMP) out[pos] = j;
    cnt += __popc(m);
    if (cnt >= NSAMP) break;
  }
  if (cnt < NSAMP) {
    const int first = (cnt > 0) ? out[0] : 0;
    for (int k = cnt + lane; k < NSAMP; k += 32) out[k] = first;
  }
}

// ---------------------------------------------------------------------------
// 3) Weight prep: W (cin x cout, f32) -> W^T padded (cout x kpad, bf16)
// ---------------------------------------------------------------------------
__global__ __launch_bounds__(256) void prep_w_kernel(const float* __restrict__ W,
                                                     __bf16* __restrict__ wT,
                                                     int cin, int cout, int kpad) {
  const int i = blockIdx.x * 256 + threadIdx.x;
  if (i < cout * kpad) {
    const int n = i / kpad;
    const int k = i - n * kpad;
    const float v = (k < cin) ? W[k * cout + n] : 0.0f;
    wT[i] = (__bf16)v;
  }
}

// ---------------------------------------------------------------------------
// 4) Grouped MLP via V_WMMA_F32_16X16X32_BF16.
//    A frag (LDS act, row-major stride K): lanes 0-15 row M, K {0..7,16..23};
//    lanes 16-31 same row, K {8..15,24..31} (ISA 7.12.2 16-bit A table).
//    B frag (global W^T, stride K): lanes 0-15 col N, K {0..7,16..23};
//    lanes 16-31 K {8..15,24..31} (same K order as A).
//    C/D f32: VGPR r = row (r + 8*half), col = lane&15.
// ---------------------------------------------------------------------------
template <int M, int K, int CO, bool FINAL>
__device__ __forceinline__ void mlp_layer(const __bf16* actIn, const __bf16* __restrict__ wT,
                                          const float* __restrict__ bias, __bf16* actOut,
                                          float* maxbuf, int tid) {
  const int lane = tid & 31;
  const int wave = tid >> 5;
  const int laneLo = lane & 15;
  const int halfSel = lane >> 4;
  constexpr int nTilesN = CO / 16;
  constexpr int nTiles = (M / 16) * nTilesN;
  for (int t = wave; t < nTiles; t += 8) {
    const int mt = t / nTilesN;
    const int nt = t - mt * nTilesN;
    const int row = mt * 16 + laneLo;
    const int col = nt * 16 + laneLo;
    f32x8 acc = {0.f, 0.f, 0.f, 0.f, 0.f, 0.f, 0.f, 0.f};
#pragma unroll
    for (int ks = 0; ks < K; ks += 32) {
      const __bf16* ap = actIn + row * K + ks + halfSel * 8;
      const bf16x8 a0 = *(const bf16x8*)(ap);
      const bf16x8 a1 = *(const bf16x8*)(ap + 16);
      const bf16x16 a = __builtin_shufflevector(a0, a1, 0, 1, 2, 3, 4, 5, 6, 7,
                                                8, 9, 10, 11, 12, 13, 14, 15);
      const __bf16* bp = wT + col * K + ks + halfSel * 8;
      const bf16x8 b0 = *(const bf16x8*)(bp);
      const bf16x8 b1 = *(const bf16x8*)(bp + 16);
      const bf16x16 bb = __builtin_shufflevector(b0, b1, 0, 1, 2, 3, 4, 5, 6, 7,
                                                 8, 9, 10, 11, 12, 13, 14, 15);
      acc = __builtin_amdgcn_wmma_f32_16x16x32_bf16(false, a, false, bb,
                                                    (short)0, acc, false, false);
    }
    const float bv = bias[col];
    if constexpr (FINAL) {
      float mx = 0.0f;
#pragma unroll
      for (int r = 0; r < 8; ++r) {
        float v = acc[r] + bv;
        v = v > 0.f ? v : 0.f;
        mx = v > mx ? v : mx;
      }
      atomicMax((int*)&maxbuf[col], __float_as_int(mx));  // all vals >= 0
    } else {
#pragma unroll
      for (int r = 0; r < 8; ++r) {
        float v = acc[r] + bv;
        v = v > 0.f ? v : 0.f;
        const int orow = mt * 16 + halfSel * 8 + r;
        actOut[orow * CO + col] = (__bf16)v;
      }
    }
  }
}

template <int NS, int C1, int C2, int C3, int OFF>
__global__ __launch_bounds__(256) void sa_mlp_kernel(
    const float* __restrict__ xyz, const float* __restrict__ points,
    const float* __restrict__ newxyz, const int* __restrict__ idx,
    const __bf16* __restrict__ w1, const __bf16* __restrict__ w2,
    const __bf16* __restrict__ w3, const float* __restrict__ b1,
    const float* __restrict__ b2, const float* __restrict__ b3,
    float* __restrict__ outFeat) {
  constexpr int K1 = 96;  // 67 padded to 96
  __shared__ __bf16 act0[NS * K1];
  __shared__ __bf16 act1[NS * C1];
  __shared__ float maxbuf[C3];
  const int tid = threadIdx.x;
  const int bs = blockIdx.x;
  const int b = bs >> 10;
  const int* myidx = idx + (size_t)bs * NS;

  for (int c = tid; c < C3; c += 256) maxbuf[c] = 0.0f;

  const float cx = newxyz[bs * 3 + 0];
  const float cy = newxyz[bs * 3 + 1];
  const float cz = newxyz[bs * 3 + 2];
  for (int e = tid; e < NS * K1; e += 256) {
    const int m = e / K1;
    const int c = e - m * K1;
    const int p = myidx[m];
    float v;
    if (c < NFEAT) {
      v = points[(size_t)(b * NPTS + p) * NFEAT + c];
    } else if (c < NFEAT + 3) {
      const int k = c - NFEAT;
      v = xyz[(b * NPTS + p) * 3 + k] - (k == 0 ? cx : (k == 1 ? cy : cz));
    } else {
      v = 0.0f;
    }
    act0[e] = (__bf16)v;
  }
  __syncthreads();
  mlp_layer<NS, K1, C1, false>(act0, w1, b1, act1, nullptr, tid);
  __syncthreads();
  mlp_layer<NS, C1, C2, false>(act1, w2, b2, act0, nullptr, tid);  // act0 reused
  __syncthreads();
  mlp_layer<NS, C2, C3, true>(act0, w3, b3, nullptr, maxbuf, tid);
  __syncthreads();
  for (int c = tid; c < C3; c += 256)
    outFeat[(size_t)bs * FEAT_STRIDE + OFF + c] = maxbuf[c];
}

// ---------------------------------------------------------------------------
// Host launcher
// ---------------------------------------------------------------------------
extern "C" void kernel_launch(void* const* d_in, const int* in_sizes, int n_in,
                              void* d_out, int out_size, void* d_ws, size_t ws_size,
                              hipStream_t stream) {
  (void)in_sizes; (void)n_in; (void)out_size; (void)ws_size;
  const float* xyz = (const float*)d_in[0];
  const float* pts = (const float*)d_in[1];

  // scratch layout
  char* wsb = (char*)d_ws;
  float* newxyz_ws = (float*)wsb;                       // 8*1024*3 f32 = 98304 B
  size_t off = 98304;
  const int nsamp[3] = {16, 32, 128};
  int* idxb[3];
  for (int s = 0; s < 3; ++s) { idxb[s] = (int*)(wsb + off); off += (size_t)NBATCH * NCTR * nsamp[s] * 4; }
  __bf16* wt = (__bf16*)(wsb + off);

  static const int g_cin[9]  = {67, 32, 32, 67, 64, 64, 67, 64, 96};
  static const int g_cout[9] = {32, 32, 64, 64, 64, 128, 64, 96, 128};
  static const int g_kpad[9] = {96, 32, 32, 96, 64, 64, 96, 64, 96};
  size_t woff[9];
  {
    size_t acc = 0;
    for (int l = 0; l < 9; ++l) { woff[l] = acc; acc += (size_t)g_cout[l] * g_kpad[l]; }
  }

  // 1) FPS (+ new_xyz into d_out head and scratch)
  fps_kernel<<<dim3(NBATCH), dim3(256), 0, stream>>>(xyz, newxyz_ws, (float*)d_out);

  // 2) weight prep (independent of FPS)
  for (int l = 0; l < 9; ++l) {
    const int total = g_cout[l] * g_kpad[l];
    prep_w_kernel<<<dim3((total + 255) / 256), dim3(256), 0, stream>>>(
        (const float*)d_in[2 + l], wt + woff[l], g_cin[l], g_cout[l], g_kpad[l]);
  }

  // 3) ball query per scale
  qball_kernel<16><<<dim3(NBATCH * NCTR / 8), dim3(256), 0, stream>>>(xyz, newxyz_ws, idxb[0], 0.1f * 0.1f);
  qball_kernel<32><<<dim3(NBATCH * NCTR / 8), dim3(256), 0, stream>>>(xyz, newxyz_ws, idxb[1], 0.2f * 0.2f);
  qball_kernel<128><<<dim3(NBATCH * NCTR / 8), dim3(256), 0, stream>>>(xyz, newxyz_ws, idxb[2], 0.4f * 0.4f);

  // 4) grouped MLP + maxpool per scale (WMMA bf16)
  float* feat = (float*)d_out + (size_t)NBATCH * NCTR * 3;
  const dim3 grid(NBATCH * NCTR), blk(256);
  sa_mlp_kernel<16, 32, 32, 64, 0><<<grid, blk, 0, stream>>>(
      xyz, pts, newxyz_ws, idxb[0], wt + woff[0], wt + woff[1], wt + woff[2],
      (const float*)d_in[11], (const float*)d_in[12], (const float*)d_in[13], feat);
  sa_mlp_kernel<32, 64, 64, 128, 64><<<grid, blk, 0, stream>>>(
      xyz, pts, newxyz_ws, idxb[1], wt + woff[3], wt + woff[4], wt + woff[5],
      (const float*)d_in[14], (const float*)d_in[15], (const float*)d_in[16], feat);
  sa_mlp_kernel<128, 64, 96, 128, 192><<<grid, blk, 0, stream>>>(
      xyz, pts, newxyz_ws, idxb[2], wt + woff[6], wt + woff[7], wt + woff[8],
      (const float*)d_in[17], (const float*)d_in[18], (const float*)d_in[19], feat);
}
